// ModulatedConv2d_1133871366294
// MI455X (gfx1250) — matrix operand order repaired and checked by reading, not compiled, not run on hardware
//
#include <hip/hip_runtime.h>

typedef _Float16 v16h __attribute__((ext_vector_type(16)));
typedef _Float16 v8h  __attribute__((ext_vector_type(8)));
typedef float    v8f  __attribute__((ext_vector_type(8)));

constexpr int Bn  = 8;
constexpr int CIN = 32;
constexpr int COUT= 32;
constexpr int H   = 512;
constexpr int W   = 512;
constexpr int MD  = 1024;
constexpr int WT  = 128;       // output pixels per block (one row segment)
constexpr int COLS= WT + 2;    // +2 halo columns

// ---------------------------------------------------------------------------
// Kernel 1: style[b][c] = dot(w[b,:], style_w[c,:]) + style_b[c]
// 8*32 = 256 outputs, each a 1024-length f32 dot. Negligible cost; one block.
// ---------------------------------------------------------------------------
__global__ __launch_bounds__(256)
void style_kernel(const float* __restrict__ w,
                  const float* __restrict__ style_w,
                  const float* __restrict__ style_b,
                  float* __restrict__ style) {
  const int t = threadIdx.x;      // 0..255
  const int b = t >> 5;
  const int c = t & 31;
  const float* wr = w + b * MD;
  const float* sr = style_w + c * MD;
  float acc = 0.f;
  for (int m = 0; m < MD; ++m) acc = fmaf(wr[m], sr[m], acc);
  style[b * CIN + c] = acc + style_b[c];
}

// ---------------------------------------------------------------------------
// Kernel 2: implicit-GEMM 3x3 conv on style-modulated activations.
// Block = 128 threads (4 waves). Each block: batch b, output row h, 128 cols.
// Each wave: 32 pixels (2 N tiles) x 32 couts (2 M tiles) -> 4 accumulators,
// 36 WMMAs; A fragments are reused across both N tiles (2.0 ds_b128/WMMA).
// ---------------------------------------------------------------------------
__global__ __launch_bounds__(128)
void conv_kernel(const float* __restrict__ x,
                 const float* __restrict__ kern,
                 const float* __restrict__ style,
                 float* __restrict__ out) {
  // Activations: [row][col][cin] f16, cin contiguous -> B fragments are
  // single contiguous 32B LDS reads per lane.
  __shared__ alignas(64) _Float16 xs[3][COLS][CIN];      // 24.4 KB
  // Weights: [tap][couttile][m][cin] f16, A-fragment-friendly.
  __shared__ alignas(64) _Float16 wl[9][2][16][CIN];     // 18 KB
  __shared__ float sstyle[CIN];                          // per-sample scales

  const int gid  = blockIdx.x;
  const int wseg = gid & 3;            // W/WT = 4 segments per row
  const int h    = (gid >> 2) & 511;
  const int b    = gid >> 11;
  const int w0   = wseg * WT;
  const int tid  = threadIdx.x;

  // ---- stage per-sample style scales ----
  if (tid < CIN) sstyle[tid] = style[b * CIN + tid];

  // ---- stage weights (9*2*16*32 = 9216 f32 -> f16), L2-resident ----
  for (int i = tid; i < 9 * 2 * 16 * CIN; i += 128) {
    const int k   = i & 31;
    const int m   = (i >> 5) & 15;
    const int t   = (i >> 9) & 1;
    const int tap = i >> 10;
    // kernel[cout][cin][kh][kw] flat, tap = kh*3+kw
    wl[tap][t][m][k] = (_Float16)kern[((t * 16 + m) * CIN + k) * 9 + tap];
  }
  __syncthreads();   // sstyle ready before activation staging reads it

  // ---- stage activation window: rows h-1..h+1, cols w0-1..w0+128 ----
  // i ordered so consecutive threads read consecutive global columns.
  for (int i = tid; i < 3 * CIN * COLS; i += 128) {
    const int c  = i % COLS;
    const int rc = i / COLS;
    const int ch = rc & 31;
    const int r  = rc >> 5;
    const int gh = h + r - 1;
    const int gw = w0 + c - 1;
    float v = 0.f;
    if ((unsigned)gh < (unsigned)H && (unsigned)gw < (unsigned)W)
      v = x[((b * CIN + ch) * H + gh) * W + gw] * sstyle[ch];
    xs[r][c][ch] = (_Float16)v;
    // Warm L2 for the block handling row h+1 (reuses 2/3 of this window).
    if (r == 2 && (c & 31) == 0 && (unsigned)(gh + 1) < (unsigned)H &&
        (unsigned)gw < (unsigned)W)
      __builtin_prefetch(&x[((b * CIN + ch) * H + gh + 1) * W + gw], 0, 1);
  }
  __syncthreads();

  const int lane = tid & 31;
  const int wv   = tid >> 5;
  const int ml   = lane & 15;
  const int hi   = lane >> 4;
  const int px0  = wv * 32 + ml;   // N tile 0 pixel
  const int px1  = px0 + 16;       // N tile 1 pixel

  v8f acc00 = {};   // cout tile 0, N tile 0
  v8f acc01 = {};   // cout tile 0, N tile 1
  v8f acc10 = {};   // cout tile 1, N tile 0
  v8f acc11 = {};   // cout tile 1, N tile 1

#pragma unroll
  for (int tap = 0; tap < 9; ++tap) {
    const int kh = tap / 3;
    const int kw = tap - kh * 3;
    // B fragments (32x16 f16): lane = pixel column, 16 consecutive channels.
    v16h b0 = *(const v16h*)&xs[kh][px0 + kw][hi * 16];
    v16h b1 = *(const v16h*)&xs[kh][px1 + kw][hi * 16];
    // A fragments (16x32 f16): lane = cout row m, 16 consecutive K per lane half.
    v8h a0lo = *(const v8h*)&wl[tap][0][ml][hi * 16];
    v8h a0hi = *(const v8h*)&wl[tap][0][ml][hi * 16 + 8];
    v8h a1lo = *(const v8h*)&wl[tap][1][ml][hi * 16];
    v8h a1hi = *(const v8h*)&wl[tap][1][ml][hi * 16 + 8];
    v16h a0 = __builtin_shufflevector(a0lo, a0hi,
        0,1,2,3,4,5,6,7,8,9,10,11,12,13,14,15);
    v16h a1 = __builtin_shufflevector(a1lo, a1hi,
        0,1,2,3,4,5,6,7,8,9,10,11,12,13,14,15);
    acc00 = __builtin_amdgcn_wmma_f32_16x16x32_f16(
        false, a0, false, b0, (short)0, acc00, false, false);
    acc01 = __builtin_amdgcn_wmma_f32_16x16x32_f16(
        false, a0, false, b1, (short)0, acc01, false, false);
    acc10 = __builtin_amdgcn_wmma_f32_16x16x32_f16(
        false, a1, false, b0, (short)0, acc10, false, false);
    acc11 = __builtin_amdgcn_wmma_f32_16x16x32_f16(
        false, a1, false, b1, (short)0, acc11, false, false);
  }

  // ---- store D: VGPR r -> cout = tile*16 + r + hi*8, N = px ----
  const int wcol0 = w0 + px0;
  const int wcol1 = w0 + px1;
#pragma unroll
  for (int r = 0; r < 8; ++r) {
    const int co = r + hi * 8;
    float* o0 = out + ((b * COUT + co) * H + h) * W;
    float* o1 = out + ((b * COUT + co + 16) * H + h) * W;
    o0[wcol0] = acc00[r];
    o0[wcol1] = acc01[r];
    o1[wcol0] = acc10[r];
    o1[wcol1] = acc11[r];
  }
}

extern "C" void kernel_launch(void* const* d_in, const int* in_sizes, int n_in,
                              void* d_out, int out_size, void* d_ws, size_t ws_size,
                              hipStream_t stream) {
  const float* x    = (const float*)d_in[0];   // [8,32,512,512]
  const float* w    = (const float*)d_in[1];   // [8,1024]
  const float* kern = (const float*)d_in[2];   // [32,32,3,3]
  const float* sw   = (const float*)d_in[3];   // [32,1024]
  const float* sb   = (const float*)d_in[4];   // [32]
  float* out   = (float*)d_out;                // [8,32,512,512]
  float* style = (float*)d_ws;                 // 256 floats scratch

  style_kernel<<<1, 256, 0, stream>>>(w, sw, sb, style);

  const int nblk = Bn * H * (W / WT);          // 8*512*4 = 16384 blocks
  conv_kernel<<<nblk, 128, 0, stream>>>(x, kern, style, out);
}
